// TestBase_47614007443728
// MI455X (gfx1250) — matrix operand order repaired and checked by reading, not compiled, not run on hardware
//
#include <hip/hip_runtime.h>
#include <stdint.h>

#define GNN_B   32
#define GNN_N   1024
#define GNN_NUM (GNN_B * GNN_N)

// Set to 0 if the async-to-LDS inline asm fails to assemble.
#ifndef GNN_USE_ASYNC
#define GNN_USE_ASYNC 1
#endif

typedef __attribute__((ext_vector_type(16))) __bf16 v16bf;
typedef __attribute__((ext_vector_type(8)))  __bf16 v8bf;
typedef __attribute__((ext_vector_type(8)))  float  v8f;

__device__ __forceinline__ unsigned short f2bf(float f) {
  unsigned int u = __float_as_uint(f);
  u += 0x7FFFu + ((u >> 16) & 1u);          // round-to-nearest-even
  return (unsigned short)(u >> 16);
}
__device__ __forceinline__ float bf2f(unsigned short h) {
  return __uint_as_float(((unsigned int)h) << 16);
}

// ---------------------------------------------------------------------------
// X0 init: nodes [NUM] f32 -> X0 [NUM x 128] bf16, col0 = value, rest zero.
// ---------------------------------------------------------------------------
__global__ void init_x0(const float* __restrict__ nodes,
                        unsigned short* __restrict__ X0) {
  int idx = blockIdx.x * blockDim.x + threadIdx.x;   // node*128 + c
  int node = idx >> 7;
  int c    = idx & 127;
  if (node >= GNN_NUM) return;
  X0[idx] = (c == 0) ? f2bf(nodes[node]) : (unsigned short)0;
}

// ---------------------------------------------------------------------------
// Weight pack: combined [Wm | Ws] (cin x Ncomb) f32 -> WMMA-B fragment order
// bf16, zero padded to Kpad x Npad. Record layout:
//   Wp[(c*Npad + n)*32 + kk] = W[k = c*32+kk][n]
// so a wave's B frag (lane l, chunk c, col tile n0) is 16 contiguous halves at
//   (c*Npad + n0 + (l&15))*32 + (l>>4)*16.
// Also emits combined bias [Npad] f32.
// ---------------------------------------------------------------------------
__global__ void pack_weights(const float* __restrict__ Wm,
                             const float* __restrict__ Ws,
                             const float* __restrict__ bm,
                             const float* __restrict__ bs,
                             unsigned short* __restrict__ Wp,
                             float* __restrict__ bias,
                             int cin, int HC, int cout, int Npad, int total) {
  int idx = blockIdx.x * blockDim.x + threadIdx.x;
  int Ncomb = HC + cout;
  if (idx < total) {
    int c   = idx / (Npad * 32);
    int rem = idx % (Npad * 32);
    int n   = rem / 32;
    int kk  = rem % 32;
    int k   = c * 32 + kk;
    float v = 0.0f;
    if (k < cin && n < Ncomb)
      v = (n < HC) ? Wm[(size_t)k * HC + n] : Ws[(size_t)k * cout + (n - HC)];
    Wp[idx] = f2bf(v);
  }
  if (idx < Npad)
    bias[idx] = (idx < HC) ? bm[idx] : (idx < Ncomb ? bs[idx - HC] : 0.0f);
}

// ---------------------------------------------------------------------------
// GEMM: Y[M x Npad] bf16 = bf16(X[M x 128]) @ Wp + bias, via
// v_wmma_f32_16x16x32_bf16.
// Block = 256 thr (8 waves) -> 64(M) x 128(N) tile; each wave owns a 64x16
// column strip with 4 accumulators, so one B fragment feeds 4 WMMAs.
// A tile (64 x Kpad bf16, <=16KB) staged to LDS with async global->LDS b128.
// ---------------------------------------------------------------------------
__global__ void gemm_wmma(const unsigned short* __restrict__ Xb,  // [M x 128] bf16
                          const unsigned short* __restrict__ Wp,  // packed B
                          const float* __restrict__ bias,         // [Npad]
                          unsigned short* __restrict__ Y,         // [M x Npad]
                          int Kpad, int Npad) {
  __shared__ unsigned short sA[64 * 128];      // 64 x Kpad bf16, row-major

  const int m0   = blockIdx.x * 64;
  const int nblk = blockIdx.y * 128;
  const int t    = threadIdx.x;

  // ---- stage A tile: 64 rows x Kpad halves, b128 async copies ----
  const int segs  = Kpad >> 3;                 // 8-half segments per row
  const int items = segs << 6;                 // 64 * segs b128 transfers
  for (int i = t; i < items; i += 256) {
    const int r = i / segs;
    const int s = i - r * segs;
    const unsigned short* g = Xb + (size_t)(m0 + r) * 128 + s * 8;
    unsigned short*       d = &sA[r * Kpad + s * 8];
#if GNN_USE_ASYNC
    unsigned int       lds = (unsigned int)(unsigned long long)(uintptr_t)d;
    unsigned long long ga  = (unsigned long long)(uintptr_t)g;
    asm volatile("global_load_async_to_lds_b128 %0, %1, off"
                 :: "v"(lds), "v"(ga) : "memory");
#else
    *(uint4*)d = *(const uint4*)g;
#endif
  }
#if GNN_USE_ASYNC
  asm volatile("s_wait_asynccnt 0" ::: "memory");
#endif
  __syncthreads();

  // ---- per-wave 64(M) x 16(N) strip: 4 accumulators share each B frag ----
  const int wave = t >> 5;
  const int lane = t & 31;
  const int n0   = nblk + wave * 16;
  const int col  = lane & 15;
  const int part = lane >> 4;
  const int arow = lane & 15;                  // A-frag row within subtile

  v8f acc[4] = {{}, {}, {}, {}};
  const int chunks = Kpad >> 5;
#pragma unroll
  for (int c = 0; c < 4; ++c) {
    if (c >= chunks) break;
    // B frag: 16 contiguous halves in packed layout (loaded once per chunk)
    const unsigned short* bptr =
        Wp + ((size_t)(c * Npad + n0 + col)) * 32 + part * 16;
    v8bf blo = *(const v8bf*)(bptr);
    v8bf bhi = *(const v8bf*)(bptr + 8);
    v16bf bfrag = __builtin_shufflevector(blo, bhi,
        0, 1, 2, 3, 4, 5, 6, 7, 8, 9, 10, 11, 12, 13, 14, 15);

#pragma unroll
    for (int mt = 0; mt < 4; ++mt) {
      // A frag: row = mt*16 + (lane&15); halves k = c*32 + part*16 + 0..15
      const unsigned short* ap =
          &sA[(mt * 16 + arow) * Kpad + c * 32 + part * 16];
      v8bf alo = *(const v8bf*)(ap);
      v8bf ahi = *(const v8bf*)(ap + 8);
      v16bf afrag = __builtin_shufflevector(alo, ahi,
          0, 1, 2, 3, 4, 5, 6, 7, 8, 9, 10, 11, 12, 13, 14, 15);

      acc[mt] = __builtin_amdgcn_wmma_f32_16x16x32_bf16(
          false, afrag, false, bfrag, (short)0, acc[mt], false, false);
    }
  }

  // ---- epilogue: D VGPR i -> (M = mt*16 + i + part*8, N = col) ----
  const int ncol = n0 + col;
  const float bv = bias[ncol];
#pragma unroll
  for (int mt = 0; mt < 4; ++mt) {
#pragma unroll
    for (int i = 0; i < 8; ++i) {
      int mrow = m0 + mt * 16 + i + part * 8;
      Y[(size_t)mrow * Npad + ncol] = f2bf(acc[mt][i] + bv);
    }
  }
}

// ---------------------------------------------------------------------------
// Attention logits: Z[node*H + h] = sum_c leaky(Y[node, h*C + c]) * att[h, c]
// ---------------------------------------------------------------------------
__global__ void attn_logits(const unsigned short* __restrict__ Y,
                            const float* __restrict__ att,
                            float* __restrict__ Z,
                            int H, int C, int Npad) {
  int idx = blockIdx.x * blockDim.x + threadIdx.x;   // node*H + h
  if (idx >= GNN_NUM * H) return;
  int node = idx / H;
  int h    = idx % H;
  const unsigned short* y = Y + (size_t)node * Npad + h * C;
  const float*          a = att + h * C;
  float s = 0.0f;
  for (int c = 0; c < C; ++c) {
    float m = bf2f(y[c]);
    m = m > 0.0f ? m : 0.01f * m;
    s += m * a[c];
  }
  Z[idx] = s;
}

// ---------------------------------------------------------------------------
// Aggregate: per (node, channel): softmax over <=2 neighbors per head,
// weighted leaky-relu messages, head mean, + self term (bias already in Y),
// optional ELU, write bf16 X_next zero-padded to 128 cols.
// ---------------------------------------------------------------------------
__global__ void aggregate(const unsigned short* __restrict__ Y,
                          const float* __restrict__ Z,
                          unsigned short* __restrict__ Xn,
                          int H, int C, int Npad, int doElu) {
  int idx  = blockIdx.x * blockDim.x + threadIdx.x;  // node*128 + c
  int node = idx >> 7;
  int c    = idx & 127;
  if (node >= GNN_NUM) return;
  if (c >= C) { Xn[idx] = 0; return; }

  int pos = node & (GNN_N - 1);
  bool hasL = pos > 0;
  bool hasR = pos < (GNN_N - 1);

  const float invH = 1.0f / (float)H;
  float acc = 0.0f;
  for (int h = 0; h < H; ++h) {
    float aL = hasL ? Z[(size_t)(node - 1) * H + h] : -1e30f;
    float aR = hasR ? Z[(size_t)(node + 1) * H + h] : -1e30f;
    float mx = fmaxf(aL, aR);
    float eL = hasL ? __expf(aL - mx) : 0.0f;
    float eR = hasR ? __expf(aR - mx) : 0.0f;
    float inv = 1.0f / (eL + eR + 1e-16f);
    float mL = 0.0f, mR = 0.0f;
    if (hasL) {
      float v = bf2f(Y[(size_t)(node - 1) * Npad + h * C + c]);
      mL = v > 0.0f ? v : 0.01f * v;
    }
    if (hasR) {
      float v = bf2f(Y[(size_t)(node + 1) * Npad + h * C + c]);
      mR = v > 0.0f ? v : 0.01f * v;
    }
    acc += (eL * mL + eR * mR) * inv;
  }
  float self = bf2f(Y[(size_t)node * Npad + H * C + c]);
  float out = acc * invH + self;
  if (doElu) out = out > 0.0f ? out : (__expf(out) - 1.0f);
  Xn[idx] = f2bf(out);
}

// ---------------------------------------------------------------------------
// Final head: per graph b, x = X5[last node][0..31];
// d_out[b] = x . Wc + bc ; d_out[32 + b*32 + c] = x[c]
// ---------------------------------------------------------------------------
__global__ void final_head(const unsigned short* __restrict__ X5,
                           const float* __restrict__ Wc,
                           const float* __restrict__ bc,
                           float* __restrict__ out) {
  int b = threadIdx.x;
  if (b >= GNN_B) return;
  const unsigned short* x = X5 + (size_t)(b * GNN_N + (GNN_N - 1)) * 128;
  float s = 0.0f;
  for (int c = 0; c < 32; ++c) {
    float v = bf2f(x[c]);
    out[32 + b * 32 + c] = v;
    s += v * Wc[c];
  }
  out[b] = s + bc[0];
}

// ---------------------------------------------------------------------------
extern "C" void kernel_launch(void* const* d_in, const int* in_sizes, int n_in,
                              void* d_out, int out_size, void* d_ws, size_t ws_size,
                              hipStream_t stream) {
  (void)in_sizes; (void)n_in; (void)out_size; (void)ws_size;
  const float* nodes = (const float*)d_in[0];

  static const int CIN[5]  = {1, 128, 128, 64, 64};
  static const int COUT[5] = {128, 128, 64, 64, 32};
  static const int HH[5]   = {8, 8, 16, 16, 1};

  char* w = (char*)d_ws;
  auto carve = [&](size_t bytes) -> char* {
    char* p = w;
    w += (bytes + 255) & ~(size_t)255;
    return p;
  };
  unsigned short* Xa   = (unsigned short*)carve((size_t)GNN_NUM * 128 * 2);
  unsigned short* Xb   = (unsigned short*)carve((size_t)GNN_NUM * 128 * 2);
  unsigned short* Y    = (unsigned short*)carve((size_t)GNN_NUM * 1152 * 2);
  float*          Z    = (float*)carve((size_t)GNN_NUM * 16 * 4);
  unsigned short* Wp   = (unsigned short*)carve((size_t)128 * 1152 * 2);
  float*          bias = (float*)carve((size_t)1152 * 4);

  init_x0<<<GNN_NUM * 128 / 256, 256, 0, stream>>>(nodes, Xa);

  unsigned short* Xc = Xa;
  unsigned short* Xn = Xb;
  for (int l = 0; l < 5; ++l) {
    const float* Wm  = (const float*)d_in[1 + l * 5 + 0];
    const float* bm  = (const float*)d_in[1 + l * 5 + 1];
    const float* Ws  = (const float*)d_in[1 + l * 5 + 2];
    const float* bs  = (const float*)d_in[1 + l * 5 + 3];
    const float* att = (const float*)d_in[1 + l * 5 + 4];

    int cin = CIN[l], cout = COUT[l], H = HH[l];
    int HC    = H * cout;
    int Ncomb = HC + cout;
    int Kpad  = (cin + 31) & ~31;
    int Npad  = (Ncomb + 127) & ~127;
    int total = Kpad * Npad;

    pack_weights<<<(total + 255) / 256, 256, 0, stream>>>(
        Wm, Ws, bm, bs, Wp, bias, cin, HC, cout, Npad, total);
    gemm_wmma<<<dim3(GNN_NUM / 64, Npad / 128), 256, 0, stream>>>(
        Xc, Wp, bias, Y, Kpad, Npad);
    attn_logits<<<(GNN_NUM * H + 255) / 256, 256, 0, stream>>>(
        Y, att, Z, H, cout, Npad);
    aggregate<<<GNN_NUM * 128 / 256, 256, 0, stream>>>(
        Y, Z, Xn, H, cout, Npad, (l < 4) ? 1 : 0);

    unsigned short* tmp = Xc; Xc = Xn; Xn = tmp;
  }

  const float* Wc = (const float*)d_in[26];
  const float* bc = (const float*)d_in[27];
  final_head<<<1, 32, 0, stream>>>(Xc, Wc, bc, (float*)d_out);
}
